// SRNN_59528246723131
// MI455X (gfx1250) — matrix operand order repaired and checked by reading, not compiled, not run on hardware
//
#include <hip/hip_runtime.h>
#include <hip/hip_bf16.h>
#include <math.h>

typedef __attribute__((ext_vector_type(16))) __bf16 v16bf;
typedef __attribute__((ext_vector_type(8)))  __bf16 v8bf;
typedef __attribute__((ext_vector_type(8)))  float  v8f;

#define H_DIM  256
#define GATE3  768
#define BATCH  512
#define NITEMS 100000
#define LAYERS 3

union V16U { v16bf v; v8bf h[2]; };

// ---------------- layer-0 x_proj: one_hot(input) @ W_ih0.T + b_ih[0] == column gather ----
__global__ void gather_xproj(const int* __restrict__ idx,
                             const float* __restrict__ W_ih0,
                             const float* __restrict__ b_ih0,
                             float* __restrict__ xproj) {
    int i = blockIdx.x * blockDim.x + threadIdx.x;
    if (i >= BATCH * GATE3) return;
    int b = i / GATE3;
    int j = i - b * GATE3;
    xproj[i] = W_ih0[(size_t)j * NITEMS + idx[b]] + b_ih0[j];
}

// ---------------- f32 -> bf16 convert (for feeding WMMA A operand) ----------------------
__global__ void cvt_bf16(const float* __restrict__ in, __bf16* __restrict__ out, int n) {
    int i = blockIdx.x * blockDim.x + threadIdx.x;
    if (i < n) out[i] = (__bf16)in[i];
}

// ---------------- C(BATCH x N) = A(BATCH x 256 bf16) @ W(N x 256 f32)^T + bias ----------
// One block = 16 output columns (items/gates) x full 512-row batch.
// W tile converted to bf16 once into LDS; 8 waves x 4 row-tiles x 8 k-steps of
// v_wmma_f32_16x16x32_bf16. W_out is therefore read from HBM exactly once.
template<bool TANH>
__launch_bounds__(256)
__global__ void gemm_wmma(const __bf16* __restrict__ A,
                          const float* __restrict__ W,
                          const float* __restrict__ bias,
                          float* __restrict__ C, int N) {
    __shared__ __bf16 sW[16 * 264];          // 16 rows x 256 bf16, +8 pad vs bank conflicts
    const int t  = threadIdx.x;
    const int n0 = blockIdx.x * 16;

    // cooperative stage + f32->bf16 convert of the 16x256 weight tile
    {
        int row = t >> 4;
        int cb  = (t & 15) * 16;
        const float* src = W + (size_t)(n0 + row) * H_DIM + cb;
        __bf16* dst = sW + row * 264 + cb;
        #pragma unroll
        for (int e = 0; e < 16; ++e) dst[e] = (__bf16)src[e];
    }
    __syncthreads();

    const int wave = t >> 5;
    const int lane = t & 31;
    const int l16  = lane & 15;
    const int kh   = lane >> 4;              // which K-half this lane holds

    v8f acc[4] = {};

    #pragma unroll
    for (int kk = 0; kk < 8; ++kk) {         // K = 256 = 8 x 32
        const int k0 = kk * 32;
        // B fragment: column l16, K-half kh -> 16 contiguous bf16 from LDS
        V16U bf;
        const v8bf* bp = (const v8bf*)(sW + l16 * 264 + k0 + kh * 16);
        bf.h[0] = bp[0];
        bf.h[1] = bp[1];
        #pragma unroll
        for (int q = 0; q < 4; ++q) {
            // A fragment: batch row, K-half kh -> 16 contiguous bf16 (K 0-15 | 16-31)
            const int row = wave * 64 + q * 16 + l16;
            const __bf16* ap = A + (size_t)row * H_DIM;
            V16U af;
            af.h[0] = *(const v8bf*)(ap + k0 + kh * 16);
            af.h[1] = *(const v8bf*)(ap + k0 + kh * 16 + 8);
            acc[q] = __builtin_amdgcn_wmma_f32_16x16x32_bf16(
                false, af.v, false, bf.v, (short)0, acc[q], false, false);
        }
    }

    const float bv = bias[n0 + l16];
    #pragma unroll
    for (int q = 0; q < 4; ++q) {
        #pragma unroll
        for (int r = 0; r < 8; ++r) {        // C/D layout: VGPR r -> M = r + 8*kh
            const int row = wave * 64 + q * 16 + kh * 8 + r;
            float v = acc[q][r] + bv;
            if (TANH) v = tanhf(v);
            C[(size_t)row * N + n0 + l16] = v;
        }
    }
}

// ---------------- fused GRU gates; writes f32 h_new (to d_out) + bf16 x for next GEMM ---
__global__ void gru_gates(const float* __restrict__ xproj,
                          const float* __restrict__ hproj,
                          const float* __restrict__ hprev,
                          float* __restrict__ hout,
                          __bf16* __restrict__ xbf) {
    int i = blockIdx.x * blockDim.x + threadIdx.x;
    if (i >= BATCH * H_DIM) return;
    int b = i >> 8;
    int j = i & 255;
    const float* xp = xproj + b * GATE3;
    const float* hp = hproj + b * GATE3;
    float r = 1.f / (1.f + expf(-(xp[j]       + hp[j])));
    float z = 1.f / (1.f + expf(-(xp[j + 256] + hp[j + 256])));
    float n = tanhf(xp[j + 512] + r * hp[j + 512]);
    float h = hprev[i];
    float hn = (1.f - z) * n + z * h;
    hout[i] = hn;
    xbf[i]  = (__bf16)hn;
}

extern "C" void kernel_launch(void* const* d_in, const int* in_sizes, int n_in,
                              void* d_out, int out_size, void* d_ws, size_t ws_size,
                              hipStream_t stream) {
    const int*   input   = (const int*)  d_in[0];
    const float* hidden  = (const float*)d_in[1];   // (L, B, H)
    const float* W_ih0   = (const float*)d_in[2];   // (3H, N)
    const float* W_ih_hi = (const float*)d_in[3];   // (L-1, 3H, H)
    const float* W_hh    = (const float*)d_in[4];   // (L, 3H, H)
    const float* b_ih    = (const float*)d_in[5];   // (L, 3H)
    const float* b_hh    = (const float*)d_in[6];   // (L, 3H)
    const float* W_out   = (const float*)d_in[7];   // (N, H)
    const float* b_out   = (const float*)d_in[8];   // (N)

    float* act  = (float*)d_out;                            // (B, N)
    float* hnew = act + (size_t)BATCH * NITEMS;             // (L, B, H)

    float*  xproj = (float*)d_ws;                           // (B, 3H)
    float*  hproj = xproj + BATCH * GATE3;                  // (B, 3H)
    __bf16* xbf   = (__bf16*)(hproj + BATCH * GATE3);       // (B, H)
    __bf16* hbf   = xbf + BATCH * H_DIM;                    // (B, H)

    const int BH = BATCH * H_DIM;
    dim3 blk(256);

    // ---- layer 0 ----
    gather_xproj<<<(BATCH * GATE3 + 255) / 256, blk, 0, stream>>>(input, W_ih0, b_ih, xproj);
    cvt_bf16<<<(BH + 255) / 256, blk, 0, stream>>>(hidden, hbf, BH);
    gemm_wmma<false><<<GATE3 / 16, blk, 0, stream>>>(hbf, W_hh, b_hh, hproj, GATE3);
    gru_gates<<<(BH + 255) / 256, blk, 0, stream>>>(xproj, hproj, hidden, hnew, xbf);

    // ---- layers 1..L-1 ----
    for (int l = 1; l < LAYERS; ++l) {
        cvt_bf16<<<(BH + 255) / 256, blk, 0, stream>>>(hidden + (size_t)l * BH, hbf, BH);
        gemm_wmma<false><<<GATE3 / 16, blk, 0, stream>>>(
            xbf, W_ih_hi + (size_t)(l - 1) * GATE3 * H_DIM, b_ih + l * GATE3, xproj, GATE3);
        gemm_wmma<false><<<GATE3 / 16, blk, 0, stream>>>(
            hbf, W_hh + (size_t)l * GATE3 * H_DIM, b_hh + l * GATE3, hproj, GATE3);
        gru_gates<<<(BH + 255) / 256, blk, 0, stream>>>(
            xproj, hproj, hidden + (size_t)l * BH, hnew + (size_t)l * BH, xbf);
    }

    // ---- output projection, HBM-bound: W_out read once, fused bias + tanh ----
    gemm_wmma<true><<<NITEMS / 16, blk, 0, stream>>>(xbf, W_out, b_out, act, NITEMS);
}
